// ConditionInjection_63282048139449
// MI455X (gfx1250) — matrix operand order repaired and checked by reading, not compiled, not run on hardware
//
#include <hip/hip_runtime.h>
#include <hip/hip_bf16.h>

// ---------------------------------------------------------------------------
// ConditionInjection for MI455X (gfx1250, wave32, WMMA)
//   - all WMMA fragments load as contiguous b128 chunks per lane
//   - flash attention K/V staged via GLOBAL_LOAD_ASYNC_TO_LDS_B128 (ASYNCcnt)
//     with double buffering: chunk t+1 DMA overlaps chunk t WMMA compute
// ---------------------------------------------------------------------------

#define DEV __device__ __forceinline__

typedef __attribute__((ext_vector_type(16))) _Float16 v16h;
typedef __attribute__((ext_vector_type(8)))  _Float16 v8h;
typedef __attribute__((ext_vector_type(8)))  float    v8f;

constexpr int C = 256, GROUPS = 32, S = 1024;     // H*W = 1024
constexpr float EPS = 1e-5f;
constexpr float QK_SCALE = 0.25f;                 // 256^-0.25, exact
constexpr float INV_SQRT2 = 0.70710678118654752f;

DEV v8f wmma32(v16h a, v16h b, v8f c) {
  return __builtin_amdgcn_wmma_f32_16x16x32_f16(false, a, false, b, (short)0, c,
                                                false, false);
}

// A fragment (16x32 f16) from row-major [M][K] source; two b128 loads per lane.
DEV v16h load_a_frag(const _Float16* base, int stride) {
  int lane = threadIdx.x & 31;
  int h = lane >> 4, r = lane & 15;
  const _Float16* p = base + r * stride + 8 * h;
  v8h lo = *(const v8h*)(p);        // K = 8h .. 8h+7       -> elems 0..7
  v8h hi = *(const v8h*)(p + 16);   // K = 16+8h .. 16+8h+7 -> elems 8..15
  return __builtin_shufflevector(lo, hi, 0, 1, 2, 3, 4, 5, 6, 7,
                                 8, 9, 10, 11, 12, 13, 14, 15);
}

// B fragment (32x16 f16) where source is [N][K] row-major (element = src[n][k]);
// lane n reads K = 8h..8h+7 and 16+8h..16+8h+7: two b128 loads.
DEV v16h load_bT_frag(const _Float16* base, int stride) {
  int lane = threadIdx.x & 31;
  int h = lane >> 4, n = lane & 15;
  const _Float16* p = base + n * stride + 8 * h;
  v8h lo = *(const v8h*)(p);
  v8h hi = *(const v8h*)(p + 16);
  return __builtin_shufflevector(lo, hi, 0, 1, 2, 3, 4, 5, 6, 7,
                                 8, 9, 10, 11, 12, 13, 14, 15);
}

// --- CDNA5 async global->LDS DMA (ASYNCcnt), 16B per lane --------------------
DEV unsigned lds_addr32(const void* p) {          // flat addr low 32 = LDS offset
  return (unsigned)(size_t)p;
}
DEV void async_copy_b128(unsigned lds_byte_addr, const _Float16* gsrc) {
  asm volatile("global_load_async_to_lds_b128 %0, %1, off"
               :: "v"(lds_byte_addr), "v"(gsrc) : "memory");
}
DEV void wait_async_0() { asm volatile("s_wait_asynccnt 0x0" ::: "memory"); }

// ---------------------------------------------------------------------------
// 1) 4x4 max-pool + SiLU + q projection (K=2, scalar); folds QK scale into q
// ---------------------------------------------------------------------------
__global__ void k_pool_q(const float* __restrict__ cond,
                         const float* __restrict__ fp2w,
                         const float* __restrict__ fp2b,
                         _Float16* __restrict__ q, int B) {
  int gid = blockIdx.x * blockDim.x + threadIdx.x;
  if (gid >= B * S) return;
  int b = gid >> 10, s = gid & 1023;
  int sy = s >> 5, sx = s & 31;
  float qo[2];
#pragma unroll
  for (int d = 0; d < 2; ++d) {
    const float* p = cond + ((size_t)(b * 2 + d) * 128 + sy * 4) * 128 + sx * 4;
    float m = -1e30f;
#pragma unroll
    for (int r = 0; r < 4; ++r) {
      float4 v4 = *(const float4*)(p + r * 128);
      m = fmaxf(m, fmaxf(fmaxf(v4.x, v4.y), fmaxf(v4.z, v4.w)));
    }
    qo[d] = m / (1.f + __expf(-m));   // SiLU
  }
  _Float16* qp = q + (size_t)(b * S + s) * C;
  for (int c0 = 0; c0 < C; c0 += 8) {
    v8h hv;
#pragma unroll
    for (int e = 0; e < 8; ++e) {
      int c = c0 + e;
      float val = qo[0] * fp2w[c * 2] + qo[1] * fp2w[c * 2 + 1] + fp2b[c];
      hv[e] = (_Float16)(val * QK_SCALE);
    }
    *(v8h*)(qp + c0) = hv;
  }
}

// ---------------------------------------------------------------------------
// 2) f32 -> f16 convert (layout preserved; weights stay [N][K])
// ---------------------------------------------------------------------------
__global__ void k_cvt_f16(const float* __restrict__ src,
                          _Float16* __restrict__ dst, int nelem) {
  int i = (blockIdx.x * blockDim.x + threadIdx.x) * 8;
  if (i >= nelem) return;
  float4 a = *(const float4*)(src + i);
  float4 b = *(const float4*)(src + i + 4);
  v8h h;
  h[0] = (_Float16)a.x; h[1] = (_Float16)a.y;
  h[2] = (_Float16)a.z; h[3] = (_Float16)a.w;
  h[4] = (_Float16)b.x; h[5] = (_Float16)b.y;
  h[6] = (_Float16)b.z; h[7] = (_Float16)b.w;
  *(v8h*)(dst + i) = h;
}

// ---------------------------------------------------------------------------
// 3) GroupNorm statistics: one block per (b, group); 8 ch x 1024 px contiguous
// ---------------------------------------------------------------------------
__global__ void k_gn_stats(const float* __restrict__ x, float* __restrict__ stats) {
  int bg = blockIdx.x;
  const float4* xp4 = (const float4*)(x + (size_t)bg * 8 * S);
  float s1 = 0.f, s2 = 0.f;
  for (int i = threadIdx.x; i < 2 * S; i += 256) {   // 8*S floats = 2*S float4
    float4 v = xp4[i];
    s1 += v.x + v.y + v.z + v.w;
    s2 += v.x * v.x + v.y * v.y + v.z * v.z + v.w * v.w;
  }
  __shared__ float sh1[256], sh2[256];
  sh1[threadIdx.x] = s1; sh2[threadIdx.x] = s2;
  __syncthreads();
  for (int st = 128; st > 0; st >>= 1) {
    if (threadIdx.x < st) {
      sh1[threadIdx.x] += sh1[threadIdx.x + st];
      sh2[threadIdx.x] += sh2[threadIdx.x + st];
    }
    __syncthreads();
  }
  if (threadIdx.x == 0) {
    float mean = sh1[0] / (8.f * S);
    float var  = sh2[0] / (8.f * S) - mean * mean;
    stats[bg * 2]     = mean;
    stats[bg * 2 + 1] = rsqrtf(var + EPS);
  }
}

// ---------------------------------------------------------------------------
// 4) normalize + transpose via LDS tile: h2[b][s][c] (f16) from x[b][c][s]
// ---------------------------------------------------------------------------
constexpr int HTS = 264;   // LDS tile row stride (halfs), 528B = 16B aligned

__global__ __launch_bounds__(256)
void k_h2(const float* __restrict__ x, const float* __restrict__ stats,
          const float* __restrict__ gw, const float* __restrict__ gb,
          _Float16* __restrict__ h2) {
  __shared__ __align__(16) _Float16 tile[64 * HTS];
  int b = blockIdx.x >> 4, stile = blockIdx.x & 15;   // S/64 = 16 tiles
  int c = threadIdx.x;                                // one channel per thread
  int g = c >> 3;
  float mean = stats[(b * GROUPS + g) * 2];
  float inv  = stats[(b * GROUPS + g) * 2 + 1];
  float wv = gw[c], bv = gb[c];
  const float* xp = x + ((size_t)b * C + c) * S + stile * 64;
#pragma unroll
  for (int i = 0; i < 64; i += 4) {
    float4 v4 = *(const float4*)(xp + i);
    tile[(i + 0) * HTS + c] = (_Float16)((v4.x - mean) * inv * wv + bv);
    tile[(i + 1) * HTS + c] = (_Float16)((v4.y - mean) * inv * wv + bv);
    tile[(i + 2) * HTS + c] = (_Float16)((v4.z - mean) * inv * wv + bv);
    tile[(i + 3) * HTS + c] = (_Float16)((v4.w - mean) * inv * wv + bv);
  }
  __syncthreads();
  int row = threadIdx.x >> 2, seg = threadIdx.x & 3;  // 4 threads per row
  _Float16* dst = h2 + ((size_t)b * S + stile * 64 + row) * C + seg * 64;
  const _Float16* srcr = &tile[row * HTS + seg * 64];
#pragma unroll
  for (int i = 0; i < 64; i += 8)
    *(v8h*)(dst + i) = *(const v8h*)(srcr + i);
}

// ---------------------------------------------------------------------------
// 5) kv GEMM: kv[s,n] = sum_c h2[s,c]*fp1_w[n,c] + b
//    n<256 -> k (scaled, [B][S][C]) ; n>=256 -> v stored transposed [B][C][S]
// ---------------------------------------------------------------------------
__global__ __launch_bounds__(256)
void k_kv_gemm(const _Float16* __restrict__ h2, const _Float16* __restrict__ w1h,
               const float* __restrict__ fp1b, _Float16* __restrict__ kbuf,
               _Float16* __restrict__ vT) {
  int wave = threadIdx.x >> 5;
  int gid = blockIdx.x * 8 + wave;          // B*64*8 wave tiles
  int nt = gid & 7;
  int st = (gid >> 3) & 63;
  int b  = gid >> 9;
  int lane = threadIdx.x & 31;
  int h = lane >> 4, n = lane & 15;
  const _Float16* A = h2 + ((size_t)b * S + st * 16) * C;
  v8f acc[4];
#pragma unroll
  for (int j = 0; j < 4; ++j) acc[j] = v8f{};
#pragma unroll
  for (int kc = 0; kc < 8; ++kc) {
    v16h a = load_a_frag(A + kc * 32, C);
#pragma unroll
    for (int j = 0; j < 4; ++j) {
      // B[k][n] = fp1_w[n][k] : [N][K] row-major source
      v16h bf = load_bT_frag(w1h + (size_t)(nt * 64 + j * 16) * C + kc * 32, C);
      acc[j] = wmma32(a, bf, acc[j]);
    }
  }
  if (nt < 4) {            // k half: keep [B][S][C] (consumed N=key contiguous-K)
#pragma unroll
    for (int j = 0; j < 4; ++j) {
      int ng = nt * 64 + j * 16 + n;
      float bias = fp1b[ng];
#pragma unroll
      for (int rr = 0; rr < 8; ++rr) {
        int srow = st * 16 + rr + 8 * h;
        kbuf[((size_t)b * S + srow) * C + ng] =
            (_Float16)((acc[j][rr] + bias) * QK_SCALE);
      }
    }
  } else {                 // v half: transpose to [B][C][S]; contiguous v8h store
#pragma unroll
    for (int j = 0; j < 4; ++j) {
      int ng = nt * 64 + j * 16 + n;
      float bias = fp1b[ng + 256];       // v weights live at rows 256..511
      v8h hv;
#pragma unroll
      for (int rr = 0; rr < 8; ++rr) hv[rr] = (_Float16)(acc[j][rr] + bias);
      *(v8h*)&vT[((size_t)b * C + ng) * S + st * 16 + 8 * h] = hv;
    }
  }
}

// ---------------------------------------------------------------------------
// 6) Flash attention: block = 8 waves x 16 query rows = 128 queries.
//    K chunk (32 keys x 256c, [key][c]) and V chunk (256c x 32 keys, [c][key])
//    double-buffered in LDS via async DMA; online softmax; P transposed via
//    per-wave LDS scratch.
// ---------------------------------------------------------------------------
constexpr int LKS = 264;       // sk row stride (halfs)
constexpr int SVS = 40;        // sv row stride (halfs), 80B = 16B aligned

__global__ __launch_bounds__(256)
void k_flash(const _Float16* __restrict__ q, const _Float16* __restrict__ kbuf,
             const _Float16* __restrict__ vT, _Float16* __restrict__ ao) {
  __shared__ __align__(16) _Float16 sk[2][32 * LKS];    // [key][c]
  __shared__ __align__(16) _Float16 sv[2][256 * SVS];   // [c][key]
  __shared__ __align__(16) _Float16 sp[8][16 * 32];     // per-wave P

  int b = blockIdx.x >> 3, qb = blockIdx.x & 7;
  int wave = threadIdx.x >> 5, lane = threadIdx.x & 31;
  int h = lane >> 4, n = lane & 15;
  int qrow0 = qb * 128 + wave * 16;

  // async-stage key chunk kt (32 keys) into LDS buffer `buf`
  auto stage = [&](int kt, int buf) {
    const _Float16* kg = kbuf + ((size_t)b * S + kt * 32) * C;
#pragma unroll
    for (int i = 0; i < 4; ++i) {          // 256 thr x 4 x 16B = 32x256 halfs
      int lin = threadIdx.x * 32 + i * 8;
      int row = lin >> 8, col = lin & 255;
      async_copy_b128(lds_addr32(&sk[buf][row * LKS + col]), kg + row * C + col);
    }
    const _Float16* vg = vT + ((size_t)b * C + threadIdx.x) * S + kt * 32;
#pragma unroll
    for (int i = 0; i < 4; ++i)            // one c-row per thread, 4 x 16B
      async_copy_b128(lds_addr32(&sv[buf][threadIdx.x * SVS + i * 8]), vg + i * 8);
  };

  const _Float16* Q = q + ((size_t)b * S + qrow0) * C;
  v16h aq[8];
#pragma unroll
  for (int kc = 0; kc < 8; ++kc) aq[kc] = load_a_frag(Q + kc * 32, C);

  v8f acc[16];
#pragma unroll
  for (int i = 0; i < 16; ++i) acc[i] = v8f{};
  float mrun[8], lrun[8];
#pragma unroll
  for (int i = 0; i < 8; ++i) { mrun[i] = -1e30f; lrun[i] = 0.f; }

  stage(0, 0);                             // prologue DMA

  for (int kt = 0; kt < 32; ++kt) {
    int cur = kt & 1;
    wait_async_0();        // my DMA into buf `cur` has landed in LDS
    __syncthreads();       // visible block-wide; everyone done reading buf cur^1
    if (kt + 1 < 32) stage(kt + 1, cur ^ 1);   // overlap next DMA with compute

    // logits S(16x32) = Q * K^T ; B[k=c][n=key] = sk[key][c] -> [N][K] source
    v8f s0 = v8f{}, s1 = v8f{};
#pragma unroll
    for (int kc = 0; kc < 8; ++kc) {
      v16h b0 = load_bT_frag(&sk[cur][0 * LKS + kc * 32], LKS);
      v16h b1 = load_bT_frag(&sk[cur][16 * LKS + kc * 32], LKS);
      s0 = wmma32(aq[kc], b0, s0);
      s1 = wmma32(aq[kc], b1, s1);
    }

    // online softmax; C/D frag row = rr + 8*h, col = n (+16 for s1)
    float alpha[8];
#pragma unroll
    for (int rr = 0; rr < 8; ++rr) {
      float mx = fmaxf(s0[rr], s1[rr]);
#pragma unroll
      for (int mk = 1; mk < 16; mk <<= 1) mx = fmaxf(mx, __shfl_xor(mx, mk, 32));
      float mnew = fmaxf(mrun[rr], mx);
      float al = __expf(mrun[rr] - mnew);
      float p0 = __expf(s0[rr] - mnew);
      float p1 = __expf(s1[rr] - mnew);
      float ps = p0 + p1;
#pragma unroll
      for (int mk = 1; mk < 16; mk <<= 1) ps += __shfl_xor(ps, mk, 32);
      lrun[rr] = lrun[rr] * al + ps;
      mrun[rr] = mnew;
      alpha[rr] = al;
      int prow = rr + 8 * h;
      sp[wave][prow * 32 + n]      = (_Float16)p0;   // P transpose via LDS
      sp[wave][prow * 32 + 16 + n] = (_Float16)p1;
    }
#pragma unroll
    for (int ntile = 0; ntile < 16; ++ntile)
#pragma unroll
      for (int rr = 0; rr < 8; ++rr) acc[ntile][rr] *= alpha[rr];

    asm volatile("s_wait_dscnt 0" ::: "memory");     // per-wave LDS RAW
    v16h ap = load_a_frag(&sp[wave][0], 32);         // P as A(16x32)
#pragma unroll
    for (int ntile = 0; ntile < 16; ++ntile) {
      // B[k=key][n=c] = sv[c][key] -> [N][K] source, stride SVS
      v16h bv = load_bT_frag(&sv[cur][(ntile * 16) * SVS], SVS);
      acc[ntile] = wmma32(ap, bv, acc[ntile]);       // O += P * V
    }
  }

#pragma unroll
  for (int rr = 0; rr < 8; ++rr) {
    float inv = 1.f / lrun[rr];
    int srow = qrow0 + rr + 8 * h;
#pragma unroll
    for (int ntile = 0; ntile < 16; ++ntile)
      ao[((size_t)b * S + srow) * C + ntile * 16 + n] =
          (_Float16)(acc[ntile][rr] * inv);
  }
}

// ---------------------------------------------------------------------------
// 7) output projection + residual + /sqrt(2): out[b][c][s] f32
// ---------------------------------------------------------------------------
__global__ __launch_bounds__(256)
void k_out_gemm(const _Float16* __restrict__ ao, const _Float16* __restrict__ owh,
                const float* __restrict__ ob, const float* __restrict__ x,
                float* __restrict__ out) {
  int wave = threadIdx.x >> 5;
  int gid = blockIdx.x * 8 + wave;          // B*64*4 wave tiles
  int nt = gid & 3;
  int st = (gid >> 2) & 63;
  int b  = gid >> 8;
  int lane = threadIdx.x & 31;
  int h = lane >> 4, n = lane & 15;
  const _Float16* A = ao + ((size_t)b * S + st * 16) * C;
  v8f acc[4];
#pragma unroll
  for (int j = 0; j < 4; ++j) acc[j] = v8f{};
#pragma unroll
  for (int kc = 0; kc < 8; ++kc) {
    v16h a = load_a_frag(A + kc * 32, C);
#pragma unroll
    for (int j = 0; j < 4; ++j) {
      v16h bf = load_bT_frag(owh + (size_t)(nt * 64 + j * 16) * C + kc * 32, C);
      acc[j] = wmma32(a, bf, acc[j]);
    }
  }
#pragma unroll
  for (int j = 0; j < 4; ++j) {
    int ng = nt * 64 + j * 16 + n;
    float bias = ob[ng];
    size_t idx = ((size_t)b * C + ng) * S + st * 16 + 8 * h;  // 8 contiguous s
    float4 x0 = *(const float4*)(x + idx);
    float4 x1 = *(const float4*)(x + idx + 4);
    float4 o0, o1;
    o0.x = (acc[j][0] + bias + x0.x) * INV_SQRT2;
    o0.y = (acc[j][1] + bias + x0.y) * INV_SQRT2;
    o0.z = (acc[j][2] + bias + x0.z) * INV_SQRT2;
    o0.w = (acc[j][3] + bias + x0.w) * INV_SQRT2;
    o1.x = (acc[j][4] + bias + x1.x) * INV_SQRT2;
    o1.y = (acc[j][5] + bias + x1.y) * INV_SQRT2;
    o1.z = (acc[j][6] + bias + x1.z) * INV_SQRT2;
    o1.w = (acc[j][7] + bias + x1.w) * INV_SQRT2;
    *(float4*)(out + idx)     = o0;
    *(float4*)(out + idx + 4) = o1;
  }
}

// ---------------------------------------------------------------------------
extern "C" void kernel_launch(void* const* d_in, const int* in_sizes, int n_in,
                              void* d_out, int out_size, void* d_ws, size_t ws_size,
                              hipStream_t stream) {
  const float* x     = (const float*)d_in[0];
  const float* cond  = (const float*)d_in[1];
  const float* gn_w  = (const float*)d_in[2];
  const float* gn_b  = (const float*)d_in[3];
  const float* fp1_w = (const float*)d_in[4];
  const float* fp1_b = (const float*)d_in[5];
  const float* fp2_w = (const float*)d_in[6];
  const float* fp2_b = (const float*)d_in[7];
  const float* out_w = (const float*)d_in[8];
  const float* out_b = (const float*)d_in[9];
  float* out = (float*)d_out;

  const int B = in_sizes[0] / (C * S);      // 32

  // workspace: q,k,vT,h2,ao (f16) + f16 weights + GN stats (~81 MB)
  char* ws = (char*)d_ws;
  size_t tile = (size_t)B * S * C * sizeof(_Float16);
  _Float16* qB  = (_Float16*)(ws + 0 * tile);
  _Float16* kB  = (_Float16*)(ws + 1 * tile);
  _Float16* vTB = (_Float16*)(ws + 2 * tile);   // [B][C][S]
  _Float16* h2B = (_Float16*)(ws + 3 * tile);
  _Float16* aoB = (_Float16*)(ws + 4 * tile);
  char* p = ws + 5 * tile;
  _Float16* w1h = (_Float16*)p; p += (size_t)2 * C * C * sizeof(_Float16);
  _Float16* owh = (_Float16*)p; p += (size_t)C * C * sizeof(_Float16);
  float* stats  = (float*)p;

  k_pool_q<<<(B * S + 255) / 256, 256, 0, stream>>>(cond, fp2_w, fp2_b, qB, B);
  k_cvt_f16<<<(2 * C * C / 8 + 255) / 256, 256, 0, stream>>>(fp1_w, w1h, 2 * C * C);
  k_cvt_f16<<<(C * C / 8 + 255) / 256, 256, 0, stream>>>(out_w, owh, C * C);
  k_gn_stats<<<B * GROUPS, 256, 0, stream>>>(x, stats);
  k_h2<<<B * 16, 256, 0, stream>>>(x, stats, gn_w, gn_b, h2B);
  k_kv_gemm<<<B * 64, 256, 0, stream>>>(h2B, w1h, fp1_b, kB, vTB);
  k_flash<<<B * 8, 256, 0, stream>>>(qB, kB, vTB, aoB);
  k_out_gemm<<<B * 32, 256, 0, stream>>>(aoB, owh, out_b, x, out);
}
